// VisionBDHv2CLS_89678917141113
// MI455X (gfx1250) — matrix operand order repaired and checked by reading, not compiled, not run on hardware
//
#include <hip/hip_runtime.h>

// ---------------------------------------------------------------------------
// VisionBDHv2CLS on gfx1250 (MI455X): all matmuls via v_wmma_f32_16x16x32_f16.
// f16 GEMM operands, f32 accumulation; f32 LN/softmax/rope.
// Token dim padded 197 -> Tp=256. GEMM: 128x64 block tile, 32x32 wave tile
// (4 WMMAs / wave / K-step), A row-major + B transposed in LDS so all
// fragment gathers are contiguous ds_load_b128 reads. Ragged-N handling is
// templated out of the hot instantiation.
// ---------------------------------------------------------------------------

typedef _Float16 f16;
typedef __attribute__((ext_vector_type(8)))  _Float16 v8h;
typedef __attribute__((ext_vector_type(16))) _Float16 v16h;
typedef __attribute__((ext_vector_type(8)))  float    v8f;

#define BSZ   64
#define TV    197
#define TP    256
#define DM    768
#define NHH   12
#define NLAT  256
#define HD    3072    // NHH * NLAT
#define NPATCH 196
#define NCLS  1000

#define BM 128
#define BN 64
#define BK 32

__device__ __forceinline__ v16h cat8(v8h lo, v8h hi) {
    return __builtin_shufflevector(lo, hi, 0,1,2,3,4,5,6,7,8,9,10,11,12,13,14,15);
}

// ======================= generic batched WMMA GEMM =========================
// C[batch] = A[batch] @ B[batch] (+bias) (ReLU optional)
// A: (M,K) f16 row-major (lda), B: (K,N) f16 row-major (ldb), C: (M,N) f32.
// Requirements: A buffer rows padded to multiple of 128, K % 32 == 0,
//               lda/ldb % 8 == 0, base pointers 16B-aligned.
//               NALIGNED: N % 64 == 0 (no ragged handling compiled in).
// per-batch offsets: off = ((batch / div) % mod) * stride  (elements)
template <bool NALIGNED>
__global__ __launch_bounds__(256)
void gemm_wmma_kernel(const f16* __restrict__ A, const f16* __restrict__ B,
                      float* __restrict__ C, const float* __restrict__ bias,
                      int M, int N, int K, int lda, int ldb, int ldc,
                      long long sA, int dA, int mA,
                      long long sB, int dB, int mB,
                      long long sC, int dC, int mC,
                      int relu)
{
    __shared__ __align__(16) f16 As [BM][BK];  // A tile, row-major (128x32) 8KB
    __shared__ __align__(16) f16 BsT[BN][BK];  // B tile, transposed (n x k) 4KB

    const int batch = blockIdx.z;
    const f16*  Ab = A + (long long)((batch / dA) % mA) * sA;
    const f16*  Bb = B + (long long)((batch / dB) % mB) * sB;
    float*      Cb = C + (long long)((batch / dC) % mC) * sC;

    const int bm  = blockIdx.y * BM;
    const int bn  = blockIdx.x * BN;
    const int tid = threadIdx.x;
    const int lane = tid & 31;
    const int wave = tid >> 5;            // 0..7
    const int wm   = (wave >> 1) * 32;    // 0,32,64,96
    const int wn   = (wave & 1) * 32;     // 0,32

    // staging assignments
    const int ar  = tid >> 1;             // A row 0..127 (2 threads/row)
    const int ac  = (tid & 1) * 16;       // A k-chunk (16 halves)
    const int bk  = tid >> 3;             // B k row 0..31 (8 threads/row)
    const int bns = (tid & 7) * 8;        // B n-chunk

    // fragment lane mapping (ISA 7.12.2, 16-bit A 16x32 / B 32x16)
    const int ah  = lane >> 4;            // A: K half-select (+8)
    const int am  = lane & 15;            // A: row in 16
    const int bnl = lane & 15;            // B: column in 16
    const int bkh = lane >> 4;            // B: K half-select (+8)

    v8f acc00 = {}, acc01 = {}, acc10 = {}, acc11 = {};

    for (int k0 = 0; k0 < K; k0 += BK) {
        // ---- stage A tile: two b128 global loads + two b128 LDS stores ----
        {
            const f16* src = Ab + (long long)(bm + ar) * lda + k0 + ac;
            *(v8h*)(&As[ar][ac])     = *(const v8h*)(src);
            *(v8h*)(&As[ar][ac + 8]) = *(const v8h*)(src + 8);
            if (k0 + BK < K)
                __builtin_prefetch(src + BK, 0, 1);
        }
        // ---- stage B tile transposed: coalesced b128 load, 8 b16 scatters ----
        {
            const f16* src = Bb + (long long)(k0 + bk) * ldb + bn + bns;
            if (NALIGNED || (bn + bns + 8 <= N)) {
                v8h v = *(const v8h*)src;
                #pragma unroll
                for (int j = 0; j < 8; ++j) BsT[bns + j][bk] = v[j];
            } else {
                #pragma unroll
                for (int j = 0; j < 8; ++j)
                    BsT[bns + j][bk] = (bn + bns + j < N) ? src[j] : (f16)0.f;
            }
        }
        __syncthreads();

        // ---- fragment gathers: contiguous 16B LDS reads ----
        const v8h* arow0 = (const v8h*)(&As[wm + am][0]);       // rows wm..wm+15
        const v8h* arow1 = (const v8h*)(&As[wm + 16 + am][0]);  // rows +16
        v16h a0 = cat8(arow0[ah], arow0[2 + ah]);
        v16h a1 = cat8(arow1[ah], arow1[2 + ah]);

        const v8h* b0row = (const v8h*)(&BsT[wn + bnl][0]);
        v16h b0 = cat8(b0row[bkh], b0row[2 + bkh]);
        const v8h* b1row = (const v8h*)(&BsT[wn + 16 + bnl][0]);
        v16h b1 = cat8(b1row[bkh], b1row[2 + bkh]);

        acc00 = __builtin_amdgcn_wmma_f32_16x16x32_f16(false, a0, false, b0,
                                                       (short)0, acc00, false, false);
        acc01 = __builtin_amdgcn_wmma_f32_16x16x32_f16(false, a0, false, b1,
                                                       (short)0, acc01, false, false);
        acc10 = __builtin_amdgcn_wmma_f32_16x16x32_f16(false, a1, false, b0,
                                                       (short)0, acc10, false, false);
        acc11 = __builtin_amdgcn_wmma_f32_16x16x32_f16(false, a1, false, b1,
                                                       (short)0, acc11, false, false);
        __syncthreads();
    }

    // ---- store C: lane 0-15 -> N=lane, M=v; lanes 16-31 -> M=v+8 ----
    const int cn  = lane & 15;
    const int cmh = lane >> 4;
    const int gn0 = bn + wn + cn;
    const int gn1 = bn + wn + 16 + cn;
    const float bias0 = bias ? ((NALIGNED || gn0 < N) ? bias[gn0] : 0.f) : 0.f;
    const float bias1 = bias ? ((NALIGNED || gn1 < N) ? bias[gn1] : 0.f) : 0.f;
    #pragma unroll
    for (int v = 0; v < 8; ++v) {
        int gm0 = bm + wm + cmh * 8 + v;
        int gm1 = gm0 + 16;
        float r00 = acc00[v] + bias0, r01 = acc01[v] + bias1;
        float r10 = acc10[v] + bias0, r11 = acc11[v] + bias1;
        if (relu) {
            r00 = fmaxf(r00, 0.f); r01 = fmaxf(r01, 0.f);
            r10 = fmaxf(r10, 0.f); r11 = fmaxf(r11, 0.f);
        }
        if (gm0 < M) {
            if (NALIGNED || gn0 < N) Cb[(long long)gm0 * ldc + gn0] = r00;
            if (NALIGNED || gn1 < N) Cb[(long long)gm0 * ldc + gn1] = r01;
        }
        if (gm1 < M) {
            if (NALIGNED || gn0 < N) Cb[(long long)gm1 * ldc + gn0] = r10;
            if (NALIGNED || gn1 < N) Cb[(long long)gm1 * ldc + gn1] = r11;
        }
    }
}

// ============================ block reductions =============================
__device__ __forceinline__ float block_sum(float v, float* red) {
    int tid = threadIdx.x;
    red[tid] = v; __syncthreads();
    for (int o = 128; o > 0; o >>= 1) {
        if (tid < o) red[tid] += red[tid + o];
        __syncthreads();
    }
    float r = red[0]; __syncthreads();
    return r;
}
__device__ __forceinline__ float block_max(float v, float* red) {
    int tid = threadIdx.x;
    red[tid] = v; __syncthreads();
    for (int o = 128; o > 0; o >>= 1) {
        if (tid < o) red[tid] = fmaxf(red[tid], red[tid + o]);
        __syncthreads();
    }
    float r = red[0]; __syncthreads();
    return r;
}

// ======================== weight conversion kernels ========================
__global__ void cvt_copy_kernel(const float* __restrict__ in, f16* __restrict__ out,
                                long long n) {
    long long i = (long long)blockIdx.x * 256 + threadIdx.x;
    if (i < n) out[i] = (f16)in[i];
}
__global__ void cvt_transpose_kernel(const float* __restrict__ in, f16* __restrict__ out,
                                     int R, int C) {  // in (R,C) -> out (C,R)
    long long i = (long long)blockIdx.x * 256 + threadIdx.x;
    if (i >= (long long)R * C) return;
    int r = (int)(i / C), c = (int)(i % C);
    out[(long long)c * R + r] = (f16)in[i];
}
__global__ void cvt_encoder_kernel(const float* __restrict__ enc, f16* __restrict__ W) {
    // enc (12,768,256) -> W (768, 3072) with col = h*256+n
    long long i = (long long)blockIdx.x * 256 + threadIdx.x;
    if (i >= (long long)NHH * DM * NLAT) return;
    int n = (int)(i % NLAT);
    long long j = i / NLAT;
    int k = (int)(j % DM);
    int h = (int)(j / DM);
    W[(long long)k * HD + h * NLAT + n] = (f16)enc[i];
}

// ============================ patch embedding ==============================
__global__ void patchify_kernel(const float* __restrict__ x, f16* __restrict__ patches) {
    long long i = (long long)blockIdx.x * 256 + threadIdx.x;
    if (i >= (long long)BSZ * NPATCH * DM) return;
    int col = (int)(i % DM);
    long long row = i / DM;
    int p = (int)(row % NPATCH);
    int b = (int)(row / NPATCH);
    int hp = p / 14, wp = p % 14;
    int c = col / 256, rem = col % 256;
    int pi = rem / 16, pj = rem % 16;
    patches[i] = (f16)x[(((long long)b * 3 + c) * 224 + hp * 16 + pi) * 224
                        + wp * 16 + pj];
}
__global__ void assemble_xt_kernel(const float* __restrict__ tok,
                                   const float* __restrict__ cls,
                                   const float* __restrict__ pos,
                                   float* __restrict__ xt) {
    long long i = (long long)blockIdx.x * 256 + threadIdx.x;
    if (i >= (long long)BSZ * TP * DM) return;
    int d = (int)(i % DM);
    long long j = i / DM;
    int t = (int)(j % TP);
    int b = (int)(j / TP);
    float v;
    if (t == 0)       v = cls[d] + pos[d];
    else if (t < TV)  v = tok[((long long)b * NPATCH + (t - 1)) * DM + d]
                          + pos[(long long)t * DM + d];
    else              v = 0.f;
    xt[i] = v;
}

// ============================== LayerNorm ==================================
__global__ __launch_bounds__(256)
void ln_rows_kernel(const float* __restrict__ in, float* __restrict__ out32,
                    f16* __restrict__ out16, int cols, int Tp, int Tvalid) {
    const int row = blockIdx.x;
    const int t = row % Tp;
    const int tid = threadIdx.x;
    __shared__ float red[256];
    long long base = (long long)row * cols;
    if (t >= Tvalid) {   // hard-zero pad rows (uniform branch)
        for (int c = tid; c < cols; c += 256) {
            if (out32) out32[base + c] = 0.f;
            if (out16) out16[base + c] = (f16)0.f;
        }
        return;
    }
    const float* r = in + base;
    float s = 0.f;
    for (int c = tid; c < cols; c += 256) s += r[c];
    float mean = block_sum(s, red) / cols;
    float v = 0.f;
    for (int c = tid; c < cols; c += 256) { float d = r[c] - mean; v += d * d; }
    float rstd = rsqrtf(block_sum(v, red) / cols + 1e-5f);
    for (int c = tid; c < cols; c += 256) {
        float o = (r[c] - mean) * rstd;
        if (out32) out32[base + c] = o;
        if (out16) out16[base + c] = (f16)o;
    }
}

// ================================ RoPE =====================================
// x_sparse (b,Tp,12,256) f32 -> qr (b,h,Tp,256) f16 and qrT (b,h,256,Tp) f16
__global__ __launch_bounds__(128)
void rope_kernel(const float* __restrict__ xs, f16* __restrict__ qr,
                 f16* __restrict__ qrT) {
    long long blk = blockIdx.x;          // over B*12*Tp
    int t = (int)(blk % TP); blk /= TP;
    int h = (int)(blk % NHH);
    int b = (int)(blk / NHH);
    int p = threadIdx.x;                 // channel pair 0..127
    int n0 = p * 2, n1 = n0 + 1;
    const float* src = xs + ((long long)(b * TP + t)) * HD + h * NLAT;
    float x0 = src[n0], x1 = src[n1];
    // freq = THETA^(-n0/256), THETA = 2^16  =>  2^(-n0/16)
    float freq = exp2f(-(float)n0 * (1.0f / 16.0f));
    float ph = (float)t * freq;
    ph = (ph - floorf(ph)) * 6.28318530717958647692f;
    float cs = cosf(ph), sn = sinf(ph);
    float o0 = x0 * cs - x1 * sn;
    float o1 = x1 * cs + x0 * sn;
    long long bh = (long long)(b * NHH + h);
    f16* q = qr + bh * TP * NLAT + (long long)t * NLAT;
    q[n0] = (f16)o0; q[n1] = (f16)o1;
    f16* qt = qrT + bh * NLAT * TP;
    qt[(long long)n0 * TP + t] = (f16)o0;
    qt[(long long)n1 * TP + t] = (f16)o1;
}

// =============================== softmax ===================================
__global__ __launch_bounds__(256)
void softmax_kernel(const float* __restrict__ scores, f16* __restrict__ attn) {
    const int t = blockIdx.x % TP;
    const long long bh = blockIdx.x / TP;
    const int tid = threadIdx.x;
    __shared__ float red[256];
    long long base = (bh * TP + t) * TP;
    if (t >= TV) {       // pad row -> zeros (uniform branch)
        attn[base + tid] = (f16)0.f;
        return;
    }
    float val = (tid < TV) ? scores[base + tid] * 0.0625f : -1e30f;
    float mx = block_max(val, red);
    float e = (tid < TV) ? expf(val - mx) : 0.f;
    float sum = block_sum(e, red);
    attn[base + tid] = (f16)(e / sum);
}

// ================== x_sparse * y_sparse  -> xy (b,t,3072) ==================
__global__ void mulmix_kernel(const float* __restrict__ ys,
                              const float* __restrict__ xs,
                              f16* __restrict__ xy) {
    long long i = (long long)blockIdx.x * 256 + threadIdx.x;  // (bh,t,n)
    if (i >= (long long)BSZ * NHH * TP * NLAT) return;
    int n = (int)(i % NLAT);
    long long j = i / NLAT;
    int t = (int)(j % TP);
    long long bh = j / TP;
    int h = (int)(bh % NHH);
    int b = (int)(bh / NHH);
    long long xi = ((long long)(b * TP + t)) * HD + h * NLAT + n;
    xy[xi] = (f16)(ys[i] * xs[xi]);
}

// ==================== xt = LN(xn + LN(yraw)), pad->0 =======================
__global__ __launch_bounds__(256)
void resln_kernel(const float* __restrict__ yraw, const float* __restrict__ xn,
                  float* __restrict__ xt) {
    const int row = blockIdx.x;
    const int t = row % TP;
    const int tid = threadIdx.x;
    __shared__ float red[256];
    long long base = (long long)row * DM;
    if (t >= TV) {
        for (int c = tid; c < DM; c += 256) xt[base + c] = 0.f;
        return;
    }
    const float* yr = yraw + base;
    const float* xr = xn + base;
    float s = 0.f;
    for (int c = tid; c < DM; c += 256) s += yr[c];
    float m1 = block_sum(s, red) * (1.0f / DM);
    float v = 0.f;
    for (int c = tid; c < DM; c += 256) { float d = yr[c] - m1; v += d * d; }
    float rs1 = rsqrtf(block_sum(v, red) * (1.0f / DM) + 1e-5f);
    float z[3];
    int k = 0;
    for (int c = tid; c < DM; c += 256) z[k++] = xr[c] + (yr[c] - m1) * rs1;
    s = 0.f;
    for (int i = 0; i < k; ++i) s += z[i];
    float m2 = block_sum(s, red) * (1.0f / DM);
    v = 0.f;
    for (int i = 0; i < k; ++i) { float d = z[i] - m2; v += d * d; }
    float rs2 = rsqrtf(block_sum(v, red) * (1.0f / DM) + 1e-5f);
    k = 0;
    for (int c = tid; c < DM; c += 256) xt[base + c] = (z[k++] - m2) * rs2;
}

// ===== cls_rep = LN(xt[:,0,:]) -> f16 (padded to 128 rows for GEMM) ========
__global__ __launch_bounds__(256)
void clsln_kernel(const float* __restrict__ xt, f16* __restrict__ cls16) {
    const int b = blockIdx.x;            // 0..127 (rows 64..127 are GEMM pad)
    const int tid = threadIdx.x;
    __shared__ float red[256];
    if (b >= BSZ) {
        for (int c = tid; c < DM; c += 256) cls16[(long long)b * DM + c] = (f16)0.f;
        return;
    }
    const float* r = xt + (long long)b * TP * DM;   // row t==0
    float s = 0.f;
    for (int c = tid; c < DM; c += 256) s += r[c];
    float mean = block_sum(s, red) * (1.0f / DM);
    float v = 0.f;
    for (int c = tid; c < DM; c += 256) { float d = r[c] - mean; v += d * d; }
    float rstd = rsqrtf(block_sum(v, red) * (1.0f / DM) + 1e-5f);
    for (int c = tid; c < DM; c += 256)
        cls16[(long long)b * DM + c] = (f16)((r[c] - mean) * rstd);
}

// ================================ launch ===================================
static inline long long cdivll(long long a, long long b) { return (a + b - 1) / b; }
static inline dim3 gemm_grid(int M, int N, int batches) {
    return dim3((unsigned)((N + BN - 1) / BN), (unsigned)((M + BM - 1) / BM),
                (unsigned)batches);
}

extern "C" void kernel_launch(void* const* d_in, const int* in_sizes, int n_in,
                              void* d_out, int out_size, void* d_ws, size_t ws_size,
                              hipStream_t stream) {
    const float* x        = (const float*)d_in[0];
    const float* patch_w  = (const float*)d_in[1];
    const float* patch_b  = (const float*)d_in[2];
    const float* cls_tok  = (const float*)d_in[3];
    const float* pos      = (const float*)d_in[4];
    const float* encoder  = (const float*)d_in[5];
    const float* encoderv = (const float*)d_in[6];
    const float* decoder  = (const float*)d_in[7];
    const float* head_w   = (const float*)d_in[8];
    const float* head_b   = (const float*)d_in[9];
    float* out = (float*)d_out;

    char* wsp = (char*)d_ws;
    auto alloc = [&](size_t bytes) -> void* {
        void* p = (void*)wsp;
        wsp += (bytes + 255) & ~(size_t)255;
        return p;
    };

    const long long rowsBT  = (long long)BSZ * TP;          // 16384
    const long long rowsBHT = (long long)BSZ * NHH * TP;    // 196608
    const int BIG = 1 << 30;

    // f16 weight copies (shared across all 6 layers)
    f16* Wenc    = (f16*)alloc((size_t)DM * HD * 2);
    f16* Wencv   = (f16*)alloc((size_t)NHH * DM * NLAT * 2);
    f16* Wdec    = (f16*)alloc((size_t)HD * DM * 2);
    f16* WpatchT = (f16*)alloc((size_t)DM * DM * 2);
    f16* WheadT  = (f16*)alloc((size_t)DM * NCLS * 2);
    // activations
    f16*   patches = (f16*)alloc((size_t)BSZ * NPATCH * DM * 2);
    float* tok32   = (float*)alloc((size_t)BSZ * NPATCH * DM * 4);
    float* xt      = (float*)alloc((size_t)rowsBT * DM * 4);
    float* xn32    = (float*)alloc((size_t)rowsBT * DM * 4);
    f16*   xn16    = (f16*)alloc((size_t)rowsBT * DM * 2);
    float* xs32    = (float*)alloc((size_t)rowsBT * HD * 4);
    f16*   qr16    = (f16*)alloc((size_t)rowsBHT * NLAT * 2);
    f16*   qrT16   = (f16*)alloc((size_t)rowsBHT * NLAT * 2);
    float* scores  = (float*)alloc((size_t)rowsBHT * TP * 4);
    f16*   attn16  = (f16*)alloc((size_t)rowsBHT * TP * 2);
    float* ykv32   = (float*)alloc((size_t)rowsBHT * DM * 4);
    f16*   ykv16   = (f16*)alloc((size_t)rowsBHT * DM * 2);
    float* ys32    = (float*)alloc((size_t)rowsBHT * NLAT * 4);
    f16*   xy16    = (f16*)alloc((size_t)rowsBT * HD * 2);
    float* yraw32  = (float*)alloc((size_t)rowsBT * DM * 4);
    f16*   cls16   = (f16*)alloc((size_t)BM * DM * 2);      // 128 rows (pad)

    // ---- one-time weight conversion ----
    cvt_encoder_kernel<<<cdivll((long long)NHH * DM * NLAT, 256), 256, 0, stream>>>(encoder, Wenc);
    cvt_copy_kernel<<<cdivll((long long)NHH * DM * NLAT, 256), 256, 0, stream>>>(
        encoderv, Wencv, (long long)NHH * DM * NLAT);
    cvt_copy_kernel<<<cdivll((long long)HD * DM, 256), 256, 0, stream>>>(
        decoder, Wdec, (long long)HD * DM);
    cvt_transpose_kernel<<<cdivll((long long)DM * DM, 256), 256, 0, stream>>>(patch_w, WpatchT, DM, DM);
    cvt_transpose_kernel<<<cdivll((long long)NCLS * DM, 256), 256, 0, stream>>>(head_w, WheadT, NCLS, DM);

    // ---- patch embedding ----
    patchify_kernel<<<cdivll((long long)BSZ * NPATCH * DM, 256), 256, 0, stream>>>(x, patches);
    gemm_wmma_kernel<true><<<gemm_grid(BSZ * NPATCH, DM, 1), 256, 0, stream>>>(
        patches, WpatchT, tok32, patch_b,
        BSZ * NPATCH, DM, DM, DM, DM, DM,
        0, 1, 1, 0, 1, 1, 0, 1, 1, 0);
    assemble_xt_kernel<<<cdivll(rowsBT * DM, 256), 256, 0, stream>>>(tok32, cls_tok, pos, xt);

    // ---- 6 encoder layers (shared weights) ----
    for (int layer = 0; layer < 6; ++layer) {
        ln_rows_kernel<<<(unsigned)rowsBT, 256, 0, stream>>>(xt, xn32, xn16, DM, TP, TV);
        // x_sparse = relu(xn @ Wenc)   (16384 x 3072, K=768)
        gemm_wmma_kernel<true><<<gemm_grid((int)rowsBT, HD, 1), 256, 0, stream>>>(
            xn16, Wenc, xs32, nullptr,
            (int)rowsBT, HD, DM, DM, HD, HD,
            0, 1, 1, 0, 1, 1, 0, 1, 1, 1);
        rope_kernel<<<(unsigned)(BSZ * NHH * TP), 128, 0, stream>>>(xs32, qr16, qrT16);
        // scores = qr @ qr^T per (b,h)   (256 x 256, K=256) x 768 batches
        gemm_wmma_kernel<true><<<gemm_grid(TP, TP, BSZ * NHH), 256, 0, stream>>>(
            qr16, qrT16, scores, nullptr,
            TP, TP, NLAT, NLAT, TP, TP,
            (long long)TP * NLAT, 1, BIG,
            (long long)NLAT * TP, 1, BIG,
            (long long)TP * TP,   1, BIG, 0);
        softmax_kernel<<<(unsigned)(BSZ * NHH * TP), 256, 0, stream>>>(scores, attn16);
        // ykv = attn @ xn   (256 x 768, K=256), V shared across heads: Boff=batch/12
        gemm_wmma_kernel<true><<<gemm_grid(TP, DM, BSZ * NHH), 256, 0, stream>>>(
            attn16, xn16, ykv32, nullptr,
            TP, DM, TP, TP, DM, DM,
            (long long)TP * TP, 1, BIG,
            (long long)TP * DM, NHH, BIG,
            (long long)TP * DM, 1, BIG, 0);
        ln_rows_kernel<<<(unsigned)rowsBHT, 256, 0, stream>>>(ykv32, nullptr, ykv16, DM, TP, TV);
        // y_sparse = relu(ykv_ln @ encoder_v[h])  (256 x 256, K=768), Boff=batch%12
        gemm_wmma_kernel<true><<<gemm_grid(TP, NLAT, BSZ * NHH), 256, 0, stream>>>(
            ykv16, Wencv, ys32, nullptr,
            TP, NLAT, DM, DM, NLAT, NLAT,
            (long long)TP * DM, 1, BIG,
            (long long)DM * NLAT, 1, NHH,
            (long long)TP * NLAT, 1, BIG, 1);
        mulmix_kernel<<<cdivll(rowsBHT * NLAT, 256), 256, 0, stream>>>(ys32, xs32, xy16);
        // y = xy @ decoder   (16384 x 768, K=3072)
        gemm_wmma_kernel<true><<<gemm_grid((int)rowsBT, DM, 1), 256, 0, stream>>>(
            xy16, Wdec, yraw32, nullptr,
            (int)rowsBT, DM, HD, HD, DM, DM,
            0, 1, 1, 0, 1, 1, 0, 1, 1, 0);
        resln_kernel<<<(unsigned)rowsBT, 256, 0, stream>>>(yraw32, xn32, xt);
    }

    // ---- classifier head (cls16 padded to 128 rows; ragged N=1000) ----
    clsln_kernel<<<BM, 256, 0, stream>>>(xt, cls16);
    gemm_wmma_kernel<false><<<gemm_grid(BSZ, NCLS, 1), 256, 0, stream>>>(
        cls16, WheadT, out, head_b,
        BSZ, NCLS, DM, DM, NCLS, NCLS,
        0, 1, 1, 0, 1, 1, 0, 1, 1, 0);
}
